// AttentionModel_87351044866223
// MI455X (gfx1250) — matrix operand order repaired and checked by reading, not compiled, not run on hardware
//
#include <hip/hip_runtime.h>

typedef __attribute__((ext_vector_type(16))) _Float16 v16h;
typedef __attribute__((ext_vector_type(8)))  _Float16 v8h;
typedef __attribute__((ext_vector_type(4)))  _Float16 v4h;
typedef __attribute__((ext_vector_type(8)))  float    v8f;

#define E_    128
#define M_    8
#define DK_   16
#define BB_   256
#define NODE_ 100
#define ROWS_ (BB_*NODE_)   // 25600
#define DFF_  512
#define EPS_  1e-5f
#define NPAD_ 112
#define NT_   7

// GEMM block tiling
#define GR_   128           // rows per block
#define GC_   64            // cols per block
#define GK_   32            // k-chunk (one WMMA K)
#define LDA_  40            // padded f16 LDS stride (16B-aligned, bank-skewed)

// ---------------------------------------------------------------- embedding
__global__ void embed_kernel(const float* __restrict__ s, const int* __restrict__ d,
                             const float* __restrict__ Wemb, const float* __restrict__ bemb,
                             const float* __restrict__ Wemb_p, const float* __restrict__ bemb_p,
                             float* __restrict__ node) {
  int gid = blockIdx.x * blockDim.x + threadIdx.x;
  if (gid >= ROWS_ * E_) return;
  int row = gid >> 7, e = gid & 127;
  int b = row / NODE_, n = row % NODE_;
  float val;
  if (n == 0) {
    float s0 = s[(size_t)b * NODE_ * 2 + 0];
    float s1 = s[(size_t)b * NODE_ * 2 + 1];
    val = s0 * Wemb_p[e] + s1 * Wemb_p[E_ + e] + bemb_p[e];
  } else {
    const float* sp = s + ((size_t)b * NODE_ + n) * 2;
    float f2 = (float)d[(size_t)b * NODE_ + n];
    val = sp[0] * Wemb[e] + sp[1] * Wemb[E_ + e] + f2 * Wemb[2 * E_ + e] + bemb[e];
  }
  node[gid] = val;
}

// ------------------------------------------- repack [M,E,DK] -> [E, M*DK]
__global__ void repack_qkv_kernel(const float* __restrict__ in, float* __restrict__ out) {
  int o = blockIdx.x * blockDim.x + threadIdx.x;
  if (o >= E_ * E_) return;
  int e = o >> 7, c = o & 127, m = c >> 4, dk = c & 15;
  out[o] = in[((size_t)m * E_ + e) * DK_ + dk];
}

// --------------------------------------------------------- LDS-staged WMMA GEMM
// Y[rows,N] = act( X[rows,K] @ W[K,N] + bias ).
// rows%128==0, K%32==0, N%64==0. Block: 256 threads = 8 waves, 128x64 tile.
// Wave w owns rows w*16..w*16+15 and all 4 column tiles (4 WMMAs per chunk).
__global__ void __launch_bounds__(256) gemm_wmma_kernel(
    const float* __restrict__ X, const float* __restrict__ W,
    const float* __restrict__ bias, float* __restrict__ Y,
    int rows, int K, int N, int relu) {
  __shared__ __align__(16) _Float16 Ash[GR_ * LDA_];   // [row][k]    128x32 (+pad)
  __shared__ __align__(16) _Float16 Bsh[GC_ * LDA_];   // [n][k]      64x32 transposed (+pad)

  int tid  = threadIdx.x;
  int lane = tid & 31, w = tid >> 5;
  int m = lane & 15, half = lane >> 4;

  int cblocks = N / GC_;
  int rb = blockIdx.x / cblocks;
  int cb = blockIdx.x % cblocks;
  int row0 = rb * GR_;
  int c0   = cb * GC_;

  v8f acc0 = {}, acc1 = {}, acc2 = {}, acc3 = {};

  for (int k0 = 0; k0 < K; k0 += GK_) {
    // ---- issue all global loads first (independent dests -> load clause)
    float4 xa[4];
#pragma unroll
    for (int u = 0; u < 4; ++u) {
      int flat = (tid + 256 * u) * 4;     // 4096 A elements, float4 granules
      int r = flat >> 5, c = flat & 31;
      xa[u] = *(const float4*)(X + (size_t)(row0 + r) * K + k0 + c);
    }
    float4 wb[2];
#pragma unroll
    for (int u = 0; u < 2; ++u) {
      int flat = (tid + 256 * u) * 4;     // 2048 B elements, float4 along n
      int kk = flat >> 6, n = flat & 63;
      wb[u] = *(const float4*)(W + (size_t)(k0 + kk) * N + c0 + n);
    }
    if (k0 + GK_ < K) {                   // prefetch next chunk -> global_prefetch_b8
      __builtin_prefetch(X + (size_t)(row0 + (tid >> 1)) * K + k0 + GK_ + (tid & 1) * 16, 0, 1);
      __builtin_prefetch(W + (size_t)(k0 + GK_ + (tid >> 6)) * N + c0 + (tid & 63), 0, 1);
    }
    // ---- stage A: f32 -> f16, packed 8B LDS stores
#pragma unroll
    for (int u = 0; u < 4; ++u) {
      int flat = (tid + 256 * u) * 4;
      int r = flat >> 5, c = flat & 31;
      v4h pk;
      pk[0] = (_Float16)xa[u].x; pk[1] = (_Float16)xa[u].y;
      pk[2] = (_Float16)xa[u].z; pk[3] = (_Float16)xa[u].w;
      *(v4h*)(Ash + r * LDA_ + c) = pk;
    }
    // ---- stage B transposed: Bsh[n][k] = W[k0+k][c0+n]
#pragma unroll
    for (int u = 0; u < 2; ++u) {
      int flat = (tid + 256 * u) * 4;
      int kk = flat >> 6, n = flat & 63;
      Bsh[(n + 0) * LDA_ + kk] = (_Float16)wb[u].x;
      Bsh[(n + 1) * LDA_ + kk] = (_Float16)wb[u].y;
      Bsh[(n + 2) * LDA_ + kk] = (_Float16)wb[u].z;
      Bsh[(n + 3) * LDA_ + kk] = (_Float16)wb[u].w;
    }
    __syncthreads();

    // ---- A fragment (ISA 16-bit A 16x32 layout), 2x contiguous 16B LDS reads
    v16h a;
    {
      const v8h* p0 = (const v8h*)(Ash + (w * 16 + m) * LDA_ + 8 * half);
      const v8h* p1 = (const v8h*)(Ash + (w * 16 + m) * LDA_ + 16 + 8 * half);
      v8h lo = p0[0], hi = p1[0];
#pragma unroll
      for (int i = 0; i < 8; ++i) { a[i] = lo[i]; a[8 + i] = hi[i]; }
    }
    // ---- B fragments + WMMA per column tile
#pragma unroll
    for (int ctw = 0; ctw < 4; ++ctw) {
      const v8h* q0 = (const v8h*)(Bsh + (ctw * 16 + m) * LDA_ + 8 * half);
      const v8h* q1 = (const v8h*)(Bsh + (ctw * 16 + m) * LDA_ + 16 + 8 * half);
      v8h blo = q0[0], bhi = q1[0];
      v16h b;
#pragma unroll
      for (int i = 0; i < 8; ++i) { b[i] = blo[i]; b[8 + i] = bhi[i]; }
      if      (ctw == 0) acc0 = __builtin_amdgcn_wmma_f32_16x16x32_f16(false, a, false, b, (short)0, acc0, false, false);
      else if (ctw == 1) acc1 = __builtin_amdgcn_wmma_f32_16x16x32_f16(false, a, false, b, (short)0, acc1, false, false);
      else if (ctw == 2) acc2 = __builtin_amdgcn_wmma_f32_16x16x32_f16(false, a, false, b, (short)0, acc2, false, false);
      else               acc3 = __builtin_amdgcn_wmma_f32_16x16x32_f16(false, a, false, b, (short)0, acc3, false, false);
    }
    __syncthreads();
  }

  // ---- epilogue: f32 C/D layout (M = r + 8*half, N = lane&15)
#pragma unroll
  for (int ctw = 0; ctw < 4; ++ctw) {
    int n = c0 + ctw * 16 + m;
    float bv = bias ? bias[n] : 0.0f;
    v8f acc = (ctw == 0) ? acc0 : (ctw == 1) ? acc1 : (ctw == 2) ? acc2 : acc3;
#pragma unroll
    for (int r = 0; r < 8; ++r) {
      int row = row0 + w * 16 + r + 8 * half;
      float v = acc[r] + bv;
      if (relu) v = v > 0.0f ? v : 0.0f;
      Y[(size_t)row * N + n] = v;
    }
  }
}

// ------------------------------------------------- attention per (batch, head)
// Q,K,V,O: [B, NODE, M*DK], head h occupies columns h*16..h*16+15.
// 7 waves: wave it owns output rows it*16..it*16+15. Softmax over key index j.
__global__ void __launch_bounds__(224) attn_kernel(
    const float* __restrict__ Q, const float* __restrict__ Km,
    const float* __restrict__ V, float* __restrict__ O, float scale) {
  __shared__ _Float16 Ssh[NPAD_ * NPAD_];   // scaled logits
  __shared__ _Float16 Psh[NPAD_ * 128];     // softmax probs, K-padded to 128

  int bb   = blockIdx.x >> 3;
  int head = blockIdx.x & 7;
  int lane = threadIdx.x & 31;
  int it   = threadIdx.x >> 5;              // wave id = i-tile
  int m = lane & 15, half = lane >> 4;

  // ---- S = (Q K^T) * scale  (K contraction = DK=16, zero-padded to 32)
  int irow = it * 16 + m;
  v16h a;
#pragma unroll
  for (int i = 0; i < 8; ++i) {
    int dk = i + 8 * half;
    a[i] = (irow < NODE_)
               ? (_Float16)Q[((size_t)bb * NODE_ + irow) * E_ + head * DK_ + dk]
               : (_Float16)0.0f;
    a[8 + i] = (_Float16)0.0f;              // K 16..31 padding
  }
  for (int jt = 0; jt < NT_; ++jt) {
    int j = jt * 16 + m;
    v16h b;
#pragma unroll
    for (int i = 0; i < 16; ++i)            // B[k=dk][n=j]; dk>=16 -> 0
      b[i] = (i < 8 && j < NODE_)
                 ? (_Float16)Km[((size_t)bb * NODE_ + j) * E_ + head * DK_ + i + 8 * half]
                 : (_Float16)0.0f;
    v8f s = {};
    s = __builtin_amdgcn_wmma_f32_16x16x32_f16(false, a, false, b,
                                               (short)0, s, false, false);
#pragma unroll
    for (int r = 0; r < 8; ++r)
      Ssh[(it * 16 + r + 8 * half) * NPAD_ + jt * 16 + m] = (_Float16)(s[r] * scale);
  }
  __syncthreads();

  // ---- row softmax (one row per thread)
  int row = threadIdx.x;
  if (row < NPAD_) {
    if (row >= NODE_) {
      for (int j = 0; j < 128; ++j) Psh[row * 128 + j] = (_Float16)0.0f;
    } else {
      float mx = -1e30f;
      for (int j = 0; j < NODE_; ++j) {
        float v = (float)Ssh[row * NPAD_ + j];
        mx = v > mx ? v : mx;
      }
      float sum = 0.0f;
      for (int j = 0; j < NODE_; ++j) {
        float ev = __expf((float)Ssh[row * NPAD_ + j] - mx);
        sum += ev;
        Psh[row * 128 + j] = (_Float16)ev;
      }
      float inv = 1.0f / sum;
      for (int j = 0; j < NODE_; ++j)
        Psh[row * 128 + j] = (_Float16)((float)Psh[row * 128 + j] * inv);
      for (int j = NODE_; j < 128; ++j) Psh[row * 128 + j] = (_Float16)0.0f;
    }
  }
  __syncthreads();

  // ---- O = P @ V   (K = 128 padded node dim, 4 WMMA steps)
  v8f acc = {};
  for (int kc = 0; kc < 4; ++kc) {
    int k0 = kc * 32;
    v16h pa, vb;
    int prow = it * 16 + m;
#pragma unroll
    for (int i = 0; i < 8; ++i) {
      pa[i]     = Psh[prow * 128 + k0 + i + 8 * half];
      pa[8 + i] = Psh[prow * 128 + k0 + 16 + i + 8 * half];
    }
#pragma unroll
    for (int i = 0; i < 16; ++i) {
      int j = k0 + (i & 7) + 8 * half + 16 * (i >> 3);
      vb[i] = (j < NODE_)
                  ? (_Float16)V[((size_t)bb * NODE_ + j) * E_ + head * DK_ + m]
                  : (_Float16)0.0f;
    }
    acc = __builtin_amdgcn_wmma_f32_16x16x32_f16(false, pa, false, vb,
                                                 (short)0, acc, false, false);
  }
#pragma unroll
  for (int r = 0; r < 8; ++r) {
    int orow = it * 16 + r + 8 * half;
    if (orow < NODE_)
      O[((size_t)bb * NODE_ + orow) * E_ + head * DK_ + m] = acc[r];
  }
}

// ------------------------------- BN over (batch, feature) per node channel
__global__ void __launch_bounds__(256) bn_nodes_kernel(
    const float* __restrict__ X, const float* __restrict__ Y,
    const float* __restrict__ w, const float* __restrict__ bp,
    float* __restrict__ Out) {
  __shared__ float s1[256], s2[256];
  int n = blockIdx.x, tid = threadIdx.x;
  float a1 = 0.0f, a2 = 0.0f;
  for (int t = tid; t < BB_ * E_; t += 256) {
    int b = t >> 7, e = t & 127;
    size_t idx = ((size_t)b * NODE_ + n) * E_ + e;
    float v = X[idx] + Y[idx];
    a1 += v; a2 += v * v;
  }
  s1[tid] = a1; s2[tid] = a2;
  __syncthreads();
  for (int off = 128; off > 0; off >>= 1) {
    if (tid < off) { s1[tid] += s1[tid + off]; s2[tid] += s2[tid + off]; }
    __syncthreads();
  }
  float mean = s1[0] / (float)(BB_ * E_);
  float var  = s2[0] / (float)(BB_ * E_) - mean * mean;
  float inv  = rsqrtf(var + EPS_);
  for (int t = tid; t < BB_ * E_; t += 256) {
    int b = t >> 7, e = t & 127;
    size_t idx = ((size_t)b * NODE_ + n) * E_ + e;
    float v = X[idx] + Y[idx];
    Out[idx] = w[e] * (v - mean) * inv + bp[e];
  }
}

// ------------------------------- BN over (batch, node) per feature channel
__global__ void __launch_bounds__(256) bn_feat_kernel(
    const float* __restrict__ X, const float* __restrict__ Y,
    const float* __restrict__ g, const float* __restrict__ bt,
    float* __restrict__ Out) {
  __shared__ float s1[256], s2[256];
  int e = blockIdx.x, tid = threadIdx.x;
  float a1 = 0.0f, a2 = 0.0f;
  for (int t = tid; t < ROWS_; t += 256) {
    size_t idx = (size_t)t * E_ + e;
    float v = X[idx] + Y[idx];
    a1 += v; a2 += v * v;
  }
  s1[tid] = a1; s2[tid] = a2;
  __syncthreads();
  for (int off = 128; off > 0; off >>= 1) {
    if (tid < off) { s1[tid] += s1[tid + off]; s2[tid] += s2[tid + off]; }
    __syncthreads();
  }
  float mean = s1[0] / (float)ROWS_;
  float var  = s2[0] / (float)ROWS_ - mean * mean;
  float inv  = rsqrtf(var + EPS_);
  for (int t = tid; t < ROWS_; t += 256) {
    size_t idx = (size_t)t * E_ + e;
    float v = X[idx] + Y[idx];
    Out[idx] = g[e] * (v - mean) * inv + bt[e];
  }
}

// ---------------------------------------------------------------- utilities
__global__ void copy_kernel(const float* __restrict__ in, float* __restrict__ out, int n) {
  int i = blockIdx.x * blockDim.x + threadIdx.x;
  if (i < n) out[i] = in[i];
}

__global__ void mean_kernel(const float* __restrict__ x, float* __restrict__ out) {
  int gid = blockIdx.x * blockDim.x + threadIdx.x;
  if (gid >= BB_ * E_) return;
  int b = gid >> 7, e = gid & 127;
  float s = 0.0f;
  for (int n = 0; n < NODE_; ++n)
    s += x[((size_t)b * NODE_ + n) * E_ + e];
  out[gid] = s * (1.0f / (float)NODE_);
}

// ---------------------------------------------------------------- driver
extern "C" void kernel_launch(void* const* d_in, const int* in_sizes, int n_in,
                              void* d_out, int out_size, void* d_ws, size_t ws_size,
                              hipStream_t stream) {
  (void)in_sizes; (void)n_in; (void)out_size; (void)ws_size;
  const float* s      = (const float*)d_in[0];
  const int*   dd     = (const int*)d_in[1];
  const float* Wemb   = (const float*)d_in[3];
  const float* bemb   = (const float*)d_in[4];
  const float* Wemb_p = (const float*)d_in[5];
  const float* bemb_p = (const float*)d_in[6];
  const float* Wq_h   = (const float*)d_in[7];
  const float* Wk_h   = (const float*)d_in[8];
  const float* Wv_h   = (const float*)d_in[9];
  const float* Wo_h   = (const float*)d_in[10];
  const float* w_bn   = (const float*)d_in[11];
  const float* b_bn   = (const float*)d_in[12];
  const float* ff0_W  = (const float*)d_in[13];
  const float* ff0_b  = (const float*)d_in[14];
  const float* ff1_W  = (const float*)d_in[15];
  const float* ff1_b  = (const float*)d_in[16];
  const float* Wq     = (const float*)d_in[17];
  const float* bq     = (const float*)d_in[18];
  const float* Wk     = (const float*)d_in[19];
  const float* bk     = (const float*)d_in[20];
  const float* Wv     = (const float*)d_in[21];
  const float* bv     = (const float*)d_in[22];
  const float* Wo     = (const float*)d_in[23];
  const float* bo     = (const float*)d_in[24];
  const float* fwW    = (const float*)d_in[25];
  const float* fwb    = (const float*)d_in[26];
  const float* fbW    = (const float*)d_in[27];
  const float* fbb    = (const float*)d_in[28];
  const float* bn_g   = (const float*)d_in[29];
  const float* bn_b   = (const float*)d_in[30];

  const size_t SZ = (size_t)ROWS_ * E_;      // 3,276,800 floats
  float* ws   = (float*)d_ws;
  float* node = ws;
  float* h    = ws + 1 * SZ;
  float* q    = ws + 2 * SZ;
  float* k    = ws + 3 * SZ;
  float* v    = ws + 4 * SZ;
  float* t    = ws + 5 * SZ;
  float* x2   = ws + 6 * SZ;
  float* ff   = ws + 7 * SZ;                 // ROWS_*DFF_ floats
  float* wqr  = ff + (size_t)ROWS_ * DFF_;
  float* wkr  = wqr + E_ * E_;
  float* wvr  = wkr + E_ * E_;

  float* out_xc  = (float*)d_out;
  float* out_x   = out_xc + SZ;
  float* out_avg = out_x + SZ;

  auto gemm = [&](const float* X, const float* W, const float* bias, float* Y,
                  int rows, int Kdim, int Ndim, int relu) {
    int blocks = (rows / GR_) * (Ndim / GC_);
    gemm_wmma_kernel<<<blocks, 256, 0, stream>>>(X, W, bias, Y, rows, Kdim, Ndim, relu);
  };

  // ---- embedding
  embed_kernel<<<(ROWS_ * E_) / 256, 256, 0, stream>>>(s, dd, Wemb, bemb, Wemb_p, bemb_p, node);

  // ---- MHA encoder path (x_compare)
  const float* hin = node;
  for (int L = 0; L < 3; ++L) {
    repack_qkv_kernel<<<64, 256, 0, stream>>>(Wq_h + (size_t)L * 16384, wqr);
    repack_qkv_kernel<<<64, 256, 0, stream>>>(Wk_h + (size_t)L * 16384, wkr);
    repack_qkv_kernel<<<64, 256, 0, stream>>>(Wv_h + (size_t)L * 16384, wvr);
    gemm(hin, wqr, nullptr, q, ROWS_, 128, 128, 0);
    gemm(hin, wkr, nullptr, k, ROWS_, 128, 128, 0);
    gemm(hin, wvr, nullptr, v, ROWS_, 128, 128, 0);
    attn_kernel<<<BB_ * M_, 224, 0, stream>>>(q, k, v, t, 0.25f);
    gemm(t, Wo_h + (size_t)L * 16384, nullptr, q, ROWS_, 128, 128, 0);   // o-proj (sum over heads)
    bn_nodes_kernel<<<NODE_, 256, 0, stream>>>(hin, q, w_bn + L * 128, b_bn + L * 128, k);
    gemm(k, ff0_W + (size_t)L * 65536, ff0_b + L * 512, ff, ROWS_, 128, 512, 1);
    gemm(ff, ff1_W + (size_t)L * 65536, ff1_b + L * 128, v, ROWS_, 512, 128, 0);
    bn_nodes_kernel<<<NODE_, 256, 0, stream>>>(k, v, w_bn + L * 128, b_bn + L * 128, h);
    hin = h;
  }
  copy_kernel<<<(ROWS_ * E_) / 256, 256, 0, stream>>>(h, out_xc, ROWS_ * E_);

  // ---- explicit encoder blocks (x)
  const float* xc  = node;
  const float* res = node;
  for (int i = 0; i < 3; ++i) {
    gemm(xc, Wq + (size_t)i * 16384, bq + i * 128, q, ROWS_, 128, 128, 0);
    gemm(xc, Wk + (size_t)i * 16384, bk + i * 128, k, ROWS_, 128, 128, 0);
    gemm(xc, Wv + (size_t)i * 16384, bv + i * 128, v, ROWS_, 128, 128, 0);
    attn_kernel<<<BB_ * M_, 224, 0, stream>>>(q, k, v, t, 0.25f);
    gemm(t, Wo + (size_t)i * 16384, bo + i * 128, ff, ROWS_, 128, 128, 0);   // Z@Wo + bo
    bn_feat_kernel<<<E_, 256, 0, stream>>>(ff, res, bn_g + (2 * i) * 128, bn_b + (2 * i) * 128, x2);
    gemm(x2, fwW + (size_t)i * 65536, fwb + i * 512, ff, ROWS_, 128, 512, 1);
    gemm(ff, fbW + (size_t)i * 65536, fbb + i * 128, t, ROWS_, 512, 128, 0);
    bn_feat_kernel<<<E_, 256, 0, stream>>>(x2, t, bn_g + (2 * i + 1) * 128, bn_b + (2 * i + 1) * 128, h);
    xc = h; res = h;
  }
  copy_kernel<<<(ROWS_ * E_) / 256, 256, 0, stream>>>(h, out_x, ROWS_ * E_);
  mean_kernel<<<(BB_ * E_) / 256, 256, 0, stream>>>(h, out_avg);
}
